// HydraAttention_82111184765009
// MI455X (gfx1250) — matrix operand order repaired and checked by reading, not compiled, not run on hardware
//
#include <hip/hip_runtime.h>

// ---------------------------------------------------------------------------
// HydraAttention for MI455X (gfx1250, wave32, WMMA).
//   out[b,l,o] = sum_c ( qn[b,l,c] * kv_sum[b,c] ) * W[o,c] + bias[o]
//   qn = q / ||q||_row ,  kv_sum[b,c] = sum_l (k/||k||_row)[b,l,c] * v[b,l,c]
// Memory-bound: ~540MB compulsory HBM traffic -> ~23us floor @ 23.3 TB/s.
// GEMM done in f16 WMMA (f32 accum); activations are unit-normalized so f16
// inputs are numerically safe and halve re-read traffic.
// Workspace layout (requires ~71.2 MB):
//   [0, 16KB)                 : kv_ws  f32 [B][C]      (atomic accumulator)
//   [16KB, 16KB+64MB)         : qh     f16 [B][L][C]   (normalized q)
//   [.., +4MB)                : Wh     f16 [B][C][C]   (W * kv_sum[b], row-major in c)
// ---------------------------------------------------------------------------

typedef _Float16 v16h __attribute__((ext_vector_type(16)));
typedef _Float16 v8h  __attribute__((ext_vector_type(8)));
typedef _Float16 v4h  __attribute__((ext_vector_type(4)));
typedef float    v8f  __attribute__((ext_vector_type(8)));
typedef float    v4f  __attribute__((ext_vector_type(4)));

#define BB 8
#define LL 8192
#define CC 512

__device__ __forceinline__ float wave_allreduce_add(float x) {
  // wave32 butterfly reduction
  #pragma unroll
  for (int off = 16; off > 0; off >>= 1) x += __shfl_xor(x, off, 32);
  return x;
}

// ---------------------------------------------------------------------------
// Kernel 0: zero the kv accumulator (workspace is poisoned by the harness).
// ---------------------------------------------------------------------------
__global__ void k0_zero(float* __restrict__ p, int n) {
  int i = blockIdx.x * blockDim.x + threadIdx.x;
  if (i < n) p[i] = 0.0f;
}

// ---------------------------------------------------------------------------
// Kernel 1: per-row L2 norms of q,k; accumulate kv_sum; emit qh = f16(qn).
// 256 threads = 8 waves; each wave processes 16 rows; block = 128 rows.
// Lane owns channels c = seg*128 + lane*4 + j  (seg 0..3, j 0..3) -> fully
// coalesced 16B vector loads across the wave.
// ---------------------------------------------------------------------------
__global__ void k1_norm_reduce(const float* __restrict__ q,
                               const float* __restrict__ k,
                               const float* __restrict__ v,
                               float* __restrict__ kv_ws,
                               _Float16* __restrict__ qh) {
  __shared__ float red[8][CC];  // per-wave partial kv sums (16 KB)

  const int lane = threadIdx.x & 31;
  const int wave = threadIdx.x >> 5;

  const long rowBase = (long)blockIdx.x * 128 + (long)wave * 16;
  const int  b       = (int)(rowBase >> 13);  // 8192 rows per batch

  v4f acc[4] = {};

  #pragma unroll 2
  for (int r = 0; r < 16; ++r) {
    const long row = rowBase + r;
    const float* kr = k + row * CC;
    const float* vr = v + row * CC;
    const float* qr = q + row * CC;

    // ---- k row: sum of squares -> rsqrt ----
    v4f kk[4];
    float ssk = 0.0f;
    #pragma unroll
    for (int s = 0; s < 4; ++s) {
      kk[s] = *(const v4f*)(kr + s * 128 + lane * 4);
      ssk += kk[s][0]*kk[s][0] + kk[s][1]*kk[s][1] + kk[s][2]*kk[s][2] + kk[s][3]*kk[s][3];
    }
    const float rk = __frsqrt_rn(wave_allreduce_add(ssk));

    // ---- accumulate kn * v ----
    #pragma unroll
    for (int s = 0; s < 4; ++s) {
      v4f vv = *(const v4f*)(vr + s * 128 + lane * 4);
      acc[s] += kk[s] * rk * vv;
    }

    // ---- q row: normalize + emit f16 ----
    v4f qq[4];
    float ssq = 0.0f;
    #pragma unroll
    for (int s = 0; s < 4; ++s) {
      qq[s] = *(const v4f*)(qr + s * 128 + lane * 4);
      ssq += qq[s][0]*qq[s][0] + qq[s][1]*qq[s][1] + qq[s][2]*qq[s][2] + qq[s][3]*qq[s][3];
    }
    const float rq = __frsqrt_rn(wave_allreduce_add(ssq));
    #pragma unroll
    for (int s = 0; s < 4; ++s) {
      v4h h = __builtin_convertvector(qq[s] * rq, v4h);
      *(v4h*)(qh + row * CC + s * 128 + lane * 4) = h;  // 8B store
    }
  }

  // ---- cross-wave reduction in LDS, then one global atomic per channel ----
  #pragma unroll
  for (int s = 0; s < 4; ++s)
    *(v4f*)(&red[wave][s * 128 + lane * 4]) = acc[s];
  __syncthreads();

  // 256 threads, 512 channels -> 2 channels per thread
  #pragma unroll
  for (int t = 0; t < 2; ++t) {
    const int c = threadIdx.x + t * 256;
    float s = 0.0f;
    #pragma unroll
    for (int w = 0; w < 8; ++w) s += red[w][c];
    atomicAdd(&kv_ws[b * CC + c], s);
  }
}

// ---------------------------------------------------------------------------
// Kernel 1.5: Wh[b][o][c] = f16( W[o][c] * kv_sum[b][c] )
// ---------------------------------------------------------------------------
__global__ void k_wscale(const float* __restrict__ W,
                         const float* __restrict__ kv_ws,
                         _Float16* __restrict__ Wh) {
  const int b = blockIdx.y;
  const int e = (blockIdx.x * 256 + threadIdx.x) * 4;  // element index in [0, C*C)
  const int c = e & (CC - 1);
  v4f w  = *(const v4f*)(W + e);
  v4f kv = *(const v4f*)(kv_ws + b * CC + c);
  *(v4h*)(Wh + (long)b * CC * CC + e) = __builtin_convertvector(w * kv, v4h);
}

// ---------------------------------------------------------------------------
// Kernel 2: out[b] = qh[b] @ Wh[b]^T + bias  via v_wmma_f32_16x16x32_f16.
// Block: 256 thr = 8 waves, tile 128(M) x 64(N); wave tile 32x32 = 4 accums.
// Both operands are row-major along K, which exactly matches the CDNA5 f16
// fragment layouts with per-lane contiguous 16B loads (no LDS needed):
//   A 16x32 : lane m(=lane&15) holds row m, k in [h*8, h*8+8) U [16+h*8, ...)
//   B 32x16 : lane n(=lane&15) holds col n, k in [h*8, h*8+8) U [16+h*8, ...)   (h=lane>>4)
// ---------------------------------------------------------------------------
__global__ void k2_gemm(const _Float16* __restrict__ qh,
                        const _Float16* __restrict__ Wh,
                        const float* __restrict__ bias,
                        float* __restrict__ out) {
  const int lane = threadIdx.x & 31;
  const int wave = threadIdx.x >> 5;
  const int wm = wave >> 1;          // 0..3
  const int wn = wave & 1;           // 0..1
  const int b  = blockIdx.z;

  const int row0 = blockIdx.y * 128 + wm * 32;
  const int col0 = blockIdx.x * 64  + wn * 32;

  const int l16  = lane & 15;
  const int half = lane >> 4;

  const _Float16* a0p = qh + ((long)b * LL + row0 + l16) * CC;
  const _Float16* a1p = a0p + (long)16 * CC;
  const _Float16* b0p = Wh + ((long)b * CC + col0 + l16) * CC;
  const _Float16* b1p = b0p + (long)16 * CC;

  const int aoff0 = half * 8;
  const int aoff1 = 16 + half * 8;
  const int boff  = half * 8;

  v8f acc00 = {}, acc01 = {}, acc10 = {}, acc11 = {};

  #pragma unroll 4
  for (int k0 = 0; k0 < CC; k0 += 32) {
    v16h a0, a1, f0, f1;
    ((v8h*)&a0)[0] = *(const v8h*)(a0p + k0 + aoff0);
    ((v8h*)&a0)[1] = *(const v8h*)(a0p + k0 + aoff1);
    ((v8h*)&a1)[0] = *(const v8h*)(a1p + k0 + aoff0);
    ((v8h*)&a1)[1] = *(const v8h*)(a1p + k0 + aoff1);
    ((v8h*)&f0)[0] = *(const v8h*)(b0p + k0 + boff);
    ((v8h*)&f0)[1] = *(const v8h*)(b0p + k0 + boff + 16);
    ((v8h*)&f1)[0] = *(const v8h*)(b1p + k0 + boff);
    ((v8h*)&f1)[1] = *(const v8h*)(b1p + k0 + boff + 16);

    acc00 = __builtin_amdgcn_wmma_f32_16x16x32_f16(false, a0, false, f0, (short)0, acc00, false, false);
    acc10 = __builtin_amdgcn_wmma_f32_16x16x32_f16(false, a1, false, f0, (short)0, acc10, false, false);
    acc01 = __builtin_amdgcn_wmma_f32_16x16x32_f16(false, a0, false, f1, (short)0, acc01, false, false);
    acc11 = __builtin_amdgcn_wmma_f32_16x16x32_f16(false, a1, false, f1, (short)0, acc11, false, false);
  }

  const float bias0 = bias[col0 + l16];
  const float bias1 = bias[col0 + 16 + l16];

  float* o = out + (long)b * LL * CC;
  // C/D layout: VGPR r -> M = half*8 + r ; N = lane&15
  #pragma unroll
  for (int r = 0; r < 8; ++r) {
    const long m0 = row0 + half * 8 + r;
    const long m1 = m0 + 16;
    o[m0 * CC + col0 + l16]      = acc00[r] + bias0;
    o[m0 * CC + col0 + 16 + l16] = acc01[r] + bias1;
    o[m1 * CC + col0 + l16]      = acc10[r] + bias0;
    o[m1 * CC + col0 + 16 + l16] = acc11[r] + bias1;
  }
}

// ---------------------------------------------------------------------------
extern "C" void kernel_launch(void* const* d_in, const int* in_sizes, int n_in,
                              void* d_out, int out_size, void* d_ws, size_t ws_size,
                              hipStream_t stream) {
  const float* q  = (const float*)d_in[0];
  const float* k  = (const float*)d_in[1];
  const float* v  = (const float*)d_in[2];
  const float* W  = (const float*)d_in[3];
  const float* bs = (const float*)d_in[4];
  float* out = (float*)d_out;

  char* ws = (char*)d_ws;
  float*    kv_ws = (float*)ws;                                   // 16 KB
  _Float16* qh    = (_Float16*)(ws + 16384);                      // 64 MB
  _Float16* Wh    = (_Float16*)(ws + 16384 + (size_t)BB*LL*CC*2); // 4 MB
  // requires ws_size >= 16384 + BB*LL*CC*2 + BB*CC*CC*2 (~71.2 MB)

  // 0) zero kv accumulator
  k0_zero<<<dim3((BB*CC + 255)/256), dim3(256), 0, stream>>>(kv_ws, BB*CC);

  // 1) norms + kv reduction + qh emission : 65536 rows / 128 rows-per-block
  k1_norm_reduce<<<dim3((BB*LL)/128), dim3(256), 0, stream>>>(q, k, v, kv_ws, qh);

  // 1.5) Wh[b] = f16(W * kv_sum[b])
  k_wscale<<<dim3((CC*CC)/(256*4), BB), dim3(256), 0, stream>>>(W, kv_ws, Wh);

  // 2) WMMA GEMM: grid x=N tiles (fastest, for L2 reuse of A), y=M tiles, z=batch
  k2_gemm<<<dim3(CC/64, LL/128, BB), dim3(256), 0, stream>>>(qh, Wh, bs, out);
}
